// ImgContrast_23261542875588
// MI455X (gfx1250) — compile-verified, matrix operand order checked by reading, not run on hardware
//
#include <hip/hip_runtime.h>
#include <hip/hip_bf16.h>
#include <cstdint>
#include <cmath>

// ---------------------------------------------------------------------------
// Types for CDNA5 WMMA (wave32): 16x16x32 bf16 -> f32
// ---------------------------------------------------------------------------
typedef __bf16 bf16_t;
typedef __attribute__((ext_vector_type(16))) __bf16 v16bf;
typedef __attribute__((ext_vector_type(8)))  float  v8f;

#define DEVINL __device__ __forceinline__

// Monotone float<->int encoding so segment-max can use integer atomicMax.
DEVINL int   enc_f(float v) { int i = __float_as_int(v); return i < 0 ? (i ^ 0x7fffffff) : i; }
DEVINL float dec_f(int i)   { return __int_as_float(i < 0 ? (i ^ 0x7fffffff) : i); }

// ---------------------------------------------------------------------------
// Weight pack: W[K,FO] f32 (row major) -> bf16 B-fragment order
//   frag f = ktile*NT + ntile ; within frag: 32 lanes x 16 contiguous bf16
//   lane<16 : n = ntile*16+lane,    k = ktile*32 + e       (e = 0..15)
//   lane>=16: n = ntile*16+lane-16, k = ktile*32 + 16 + e
// ---------------------------------------------------------------------------
__global__ void pack_w_kernel(const float* __restrict__ W, bf16_t* __restrict__ Wp,
                              int K, int FO) {
    int i = blockIdx.x * blockDim.x + threadIdx.x;
    int total = K * FO;
    if (i >= total) return;
    int e    = i & 15;
    int lane = (i >> 4) & 31;
    int f    = i >> 9;
    int NT   = FO >> 4;
    int kt   = f / NT;
    int nt   = f - kt * NT;
    int k    = kt * 32 + ((lane >> 4) << 4) + e;
    int n    = nt * 16 + (lane & 15);
    Wp[i] = (bf16_t)W[(size_t)k * FO + n];
}

__global__ void f32_to_bf16_kernel(const float* __restrict__ x, bf16_t* __restrict__ y, int n) {
    int i = blockIdx.x * blockDim.x + threadIdx.x;
    if (i < n) y[i] = (bf16_t)x[i];
}

__global__ void elu_to_bf16_kernel(const float* __restrict__ x, bf16_t* __restrict__ y, int n) {
    int i = blockIdx.x * blockDim.x + threadIdx.x;
    if (i < n) {
        float v = x[i];
        v = v > 0.0f ? v : expm1f(v);   // jax.nn.elu, alpha = 1
        y[i] = (bf16_t)v;
    }
}

__global__ void seg_init_kernel(int* __restrict__ amax, float* __restrict__ den, int n) {
    int i = blockIdx.x * blockDim.x + threadIdx.x;
    if (i < n) { amax[i] = INT_MIN; den[i] = 0.0f; }
}

// ---------------------------------------------------------------------------
// GEMM: Y[M,FO] = X[M,K] @ W + bias
//   X bf16 row-major; W pre-packed into B fragments, staged in LDS per block
//   through the CDNA5 async memory->LDS path (GLOBAL_LOAD_ASYNC_TO_LDS_B128,
//   ASYNCcnt). Each wave owns TWO 16-row strips: one LDS-resident B fragment
//   feeds two WMMAs, doubling arithmetic intensity.
// ---------------------------------------------------------------------------
template <int K, int FO>
__global__ __launch_bounds__(256)
void gemm_bf16_wmma(const bf16_t* __restrict__ X, const bf16_t* __restrict__ Wp,
                    const float* __restrict__ bias, float* __restrict__ Y, int M) {
    constexpr int KT = K / 32;
    constexpr int NT = FO / 16;
    constexpr int M2 = 2;                       // row strips per wave

    extern __shared__ char smem_raw[];
    bf16_t* Wl = (bf16_t*)smem_raw;

    // Async-stage packed weights into LDS: direct memory->LDS DMA per lane,
    // 16B per chunk, no VGPR round trip. LDS aperture maps addr[31:0] to the
    // LDS byte offset, so truncating the generic pointer gives the DS address.
    {
        const unsigned lds_base = (unsigned)(uintptr_t)Wl;
        const int nchunk = (K * FO) / 8;        // 8 bf16 per 16B chunk
        for (int i = threadIdx.x; i < nchunk; i += blockDim.x) {
            unsigned      loff = lds_base + (unsigned)i * 16u;
            const bf16_t* g    = Wp + (size_t)i * 8;
            asm volatile("global_load_async_to_lds_b128 %0, %1, off"
                         :: "v"(loff), "v"(g) : "memory");
        }
        asm volatile("s_wait_asynccnt 0x0" ::: "memory");
    }
    __syncthreads();

    const int wave   = threadIdx.x >> 5;
    const int lane   = threadIdx.x & 31;
    const int strip0 = (blockIdx.x * (blockDim.x >> 5) + wave) * M2;
    const int m0     = strip0 << 4;
    if (m0 >= M) return;                        // wave-uniform: EXEC stays all-1s
    const bool has2  = (m0 + 16) < M;           // wave-uniform

    const int lo = lane & 15;
    const int hi = lane >> 4;
    const bf16_t* Arow0 = X + (size_t)(m0 + lo) * K + (size_t)hi * 16;
    const bf16_t* Arow1 = Arow0 + (has2 ? (size_t)16 * K : 0);  // clamped: no EXEC games

    v8f acc[M2][NT];
#pragma unroll
    for (int s = 0; s < M2; ++s)
#pragma unroll
        for (int n = 0; n < NT; ++n)
#pragma unroll
            for (int r = 0; r < 8; ++r) acc[s][n][r] = 0.0f;

#pragma unroll
    for (int kt = 0; kt < KT; ++kt) {
        v16bf a0 = *(const v16bf*)(Arow0 + (size_t)kt * 32);
        v16bf a1 = *(const v16bf*)(Arow1 + (size_t)kt * 32);
#pragma unroll
        for (int n = 0; n < NT; ++n) {
            v16bf b = *(const v16bf*)(Wl + ((size_t)(kt * NT + n) * 32 + lane) * 16);
            acc[0][n] = __builtin_amdgcn_wmma_f32_16x16x32_bf16(
                false, a0, false, b, (short)0, acc[0][n], false, false);
            acc[1][n] = __builtin_amdgcn_wmma_f32_16x16x32_bf16(
                false, a1, false, b, (short)0, acc[1][n], false, false);
        }
    }

#pragma unroll
    for (int s = 0; s < M2; ++s) {
        if (s == 1 && !has2) break;             // wave-uniform
        const int rbase = m0 + s * 16 + (hi << 3);
#pragma unroll
        for (int n = 0; n < NT; ++n) {
            const int   col = n * 16 + lo;
            const float bc  = bias[col];
#pragma unroll
            for (int r = 0; r < 8; ++r)
                Y[(size_t)(rbase + r) * FO + col] = acc[s][n][r] + bc;
        }
    }
}

// ---------------------------------------------------------------------------
// Edge phase
// ---------------------------------------------------------------------------
template <int FO>
__global__ __launch_bounds__(256)
void edge_logits_kernel(const float* __restrict__ q, const float* __restrict__ k,
                        const int* __restrict__ src, const int* __restrict__ dst,
                        float* __restrict__ logits, int* __restrict__ amax,
                        float scale, int E) {
    const int w    = blockIdx.x * (blockDim.x >> 5) + (threadIdx.x >> 5);
    const int lane = threadIdx.x & 31;
    if (w >= E) return;
    const int s = src[w], d = dst[w];
    float sum = 0.0f;
    if (lane * 4 < FO) {
        float4 a = *(const float4*)(q + (size_t)d * FO + lane * 4);
        float4 b = *(const float4*)(k + (size_t)s * FO + lane * 4);
        sum = a.x * b.x + a.y * b.y + a.z * b.z + a.w * b.w;
    }
#pragma unroll
    for (int off = 16; off >= 1; off >>= 1) sum += __shfl_xor(sum, off, 32);
    if (lane == 0) {
        float l = sum * scale;
        logits[w] = l;
        atomicMax(amax + d, enc_f(l));
    }
}

__global__ __launch_bounds__(256)
void edge_exp_kernel(float* __restrict__ logits_ex, const int* __restrict__ amax,
                     const int* __restrict__ src, const int* __restrict__ dst,
                     float* __restrict__ den, const float* __restrict__ v,
                     int FO, int E) {
    int e = blockIdx.x * blockDim.x + threadIdx.x;
    if (e >= E) return;
    int d = dst[e];
    float ex = __expf(logits_ex[e] - dec_f(amax[d]));
    logits_ex[e] = ex;                 // in place: logits -> exp
    atomicAdd(den + d, ex);
    // Warm L2 for the upcoming scatter of v[src] (global_prefetch_b8).
    __builtin_prefetch(v + (size_t)src[e] * FO, 0, 1);
}

template <int FO>
__global__ __launch_bounds__(256)
void edge_scatter_kernel(const float* __restrict__ ex, const float* __restrict__ den,
                         const float* __restrict__ v, const int* __restrict__ src,
                         const int* __restrict__ dst, float* __restrict__ out, int E) {
    const int w    = blockIdx.x * (blockDim.x >> 5) + (threadIdx.x >> 5);
    const int lane = threadIdx.x & 31;
    if (w >= E) return;
    const int s = src[w], d = dst[w];
    const float alpha = ex[w] / den[d];
    if (lane * 4 < FO) {
        float4 vv = *(const float4*)(v + (size_t)s * FO + lane * 4);
        float* o  = out + (size_t)d * FO + lane * 4;
        atomicAdd(o + 0, alpha * vv.x);
        atomicAdd(o + 1, alpha * vv.y);
        atomicAdd(o + 2, alpha * vv.z);
        atomicAdd(o + 3, alpha * vv.w);
    }
}

// ---------------------------------------------------------------------------
// Host orchestration
// ---------------------------------------------------------------------------
extern "C" void kernel_launch(void* const* d_in, const int* in_sizes, int n_in,
                              void* d_out, int out_size, void* d_ws, size_t ws_size,
                              hipStream_t stream) {
    (void)n_in; (void)out_size; (void)ws_size;
    const int N = in_sizes[0] / 256;
    const int E = in_sizes[2] / 2;
    const float* xi = (const float*)d_in[0];
    const float* xj = (const float*)d_in[1];
    const int*   src = (const int*)d_in[2];
    const int*   dst = src + E;
    float* out = (float*)d_out;

    // ---- workspace carve ----
    uintptr_t p = (uintptr_t)d_ws;
    auto carve = [&](size_t bytes) -> void* {
        p = (p + 255) & ~(uintptr_t)255;
        void* r = (void*)p;
        p += bytes;
        return r;
    };
    bf16_t* Xbf  = (bf16_t*)carve((size_t)N * 256 * 2);
    bf16_t* Hbf  = (bf16_t*)carve((size_t)N * 128 * 2);
    float*  Q    = (float*)carve((size_t)N * 128 * 4);
    float*  Kf   = (float*)carve((size_t)N * 128 * 4);
    float*  V    = (float*)carve((size_t)N * 128 * 4);
    float*  T1   = (float*)carve((size_t)N * 128 * 4);
    float*  EXb  = (float*)carve((size_t)E * 4);
    int*    AMAX = (int*)carve((size_t)N * 4);
    float*  DEN  = (float*)carve((size_t)N * 4);

    const int KD[3] = {256, 128, 128};
    const int FD[3] = {128, 128, 16};
    bf16_t* Wp[12];
    for (int l = 0; l < 3; ++l)
        for (int m = 0; m < 4; ++m)
            Wp[l * 4 + m] = (bf16_t*)carve((size_t)KD[l] * FD[l] * 2);

    // ---- pack all 12 weight matrices into WMMA B-fragment order ----
    for (int l = 0; l < 3; ++l)
        for (int m = 0; m < 4; ++m) {
            int tot = KD[l] * FD[l];
            pack_w_kernel<<<(tot + 255) / 256, 256, 0, stream>>>(
                (const float*)d_in[3 + l * 8 + m * 2], Wp[l * 4 + m], KD[l], FD[l]);
        }

    // 8 waves/block, 2 strips (32 rows) per wave -> 256 rows per block
    const int strips     = (N + 15) / 16;
    const int gemmBlocks = (strips + 15) / 16;
    const int edgeBlocks = (E + 7) / 8;                 // wave per edge
    const int edgeThB    = (E + 255) / 256;
    const int nodeB      = (N + 255) / 256;

    auto gemm = [&](int layer, const bf16_t* xb, int widx, const float* b, float* y) {
        const size_t sh = (size_t)KD[layer] * FD[layer] * 2;  // LDS staging bytes
        if (layer == 0)
            gemm_bf16_wmma<256, 128><<<gemmBlocks, 256, sh, stream>>>(xb, Wp[widx], b, y, N);
        else if (layer == 1)
            gemm_bf16_wmma<128, 128><<<gemmBlocks, 256, sh, stream>>>(xb, Wp[widx], b, y, N);
        else
            gemm_bf16_wmma<128, 16><<<gemmBlocks, 256, sh, stream>>>(xb, Wp[widx], b, y, N);
    };

    auto tconv = [&](int layer, const bf16_t* xb, float* o) {
        const int base = 3 + layer * 8;
        const int wb   = layer * 4;
        gemm(layer, xb, wb + 0, (const float*)d_in[base + 1], Q);   // q
        gemm(layer, xb, wb + 1, (const float*)d_in[base + 3], Kf);  // k
        gemm(layer, xb, wb + 2, (const float*)d_in[base + 5], V);   // v
        gemm(layer, xb, wb + 3, (const float*)d_in[base + 7], o);   // skip: o = x@sw + sb
        seg_init_kernel<<<nodeB, 256, 0, stream>>>(AMAX, DEN, N);
        const int   FO    = FD[layer];
        const float scale = 1.0f / sqrtf((float)FO);
        if (FO == 128) {
            edge_logits_kernel<128><<<edgeBlocks, 256, 0, stream>>>(Q, Kf, src, dst, EXb, AMAX, scale, E);
            edge_exp_kernel<<<edgeThB, 256, 0, stream>>>(EXb, AMAX, src, dst, DEN, V, FO, E);
            edge_scatter_kernel<128><<<edgeBlocks, 256, 0, stream>>>(EXb, DEN, V, src, dst, o, E);
        } else {
            edge_logits_kernel<16><<<edgeBlocks, 256, 0, stream>>>(Q, Kf, src, dst, EXb, AMAX, scale, E);
            edge_exp_kernel<<<edgeThB, 256, 0, stream>>>(EXb, AMAX, src, dst, DEN, V, FO, E);
            edge_scatter_kernel<16><<<edgeBlocks, 256, 0, stream>>>(EXb, DEN, V, src, dst, o, E);
        }
    };

    const size_t oHi2 = 0;
    const size_t oHj2 = (size_t)N * 128;
    const size_t oCi  = (size_t)N * 256;
    const size_t oCj  = (size_t)N * 256 + (size_t)N * 16;

    // ---------------- i-stream ----------------
    f32_to_bf16_kernel<<<(N * 256 + 255) / 256, 256, 0, stream>>>(xi, Xbf, N * 256);
    tconv(0, Xbf, T1);                                                        // hi1 (pre-ELU)
    elu_to_bf16_kernel<<<(N * 128 + 255) / 256, 256, 0, stream>>>(T1, Hbf, N * 128);
    tconv(1, Hbf, out + oHi2);                                                // hi2
    tconv(2, Hbf, out + oCi);                                                 // ci = tconv(hi1)

    // ---------------- j-stream ----------------
    f32_to_bf16_kernel<<<(N * 256 + 255) / 256, 256, 0, stream>>>(xj, Xbf, N * 256);
    tconv(0, Xbf, T1);                                                        // hj1 (pre-ELU)
    elu_to_bf16_kernel<<<(N * 128 + 255) / 256, 256, 0, stream>>>(T1, Hbf, N * 128);
    tconv(1, Hbf, out + oHj2);                                                // hj2
    f32_to_bf16_kernel<<<(N * 128 + 255) / 256, 256, 0, stream>>>(out + oHj2, Hbf, N * 128);
    tconv(2, Hbf, out + oCj);                                                 // cj = tconv(hj2)
}
